// CausalSelfAttention_21440476741875
// MI455X (gfx1250) — matrix operand order repaired and checked by reading, not compiled, not run on hardware
//
#include <hip/hip_runtime.h>

typedef __attribute__((ext_vector_type(16))) __bf16       v16bf;
typedef __attribute__((ext_vector_type(8)))  float        v8f;
typedef __attribute__((ext_vector_type(4)))  unsigned int u32x4;

#define E_   1024
#define S_   2048
#define B_   4
#define H_   16
#define D_   64
#define BS_  (B_ * S_)    // 8192
#define N3_  (3 * E_)     // 3072

union BFrag { v16bf v; u32x4 q[2]; };

__device__ __forceinline__ unsigned short f2bf(float f) {
    unsigned int u = __float_as_uint(f);
    u += 0x7fffu + ((u >> 16) & 1u);           // round-to-nearest-even
    return (unsigned short)(u >> 16);
}

__device__ __forceinline__ BFrag load_frag2(const unsigned short* p0,
                                            const unsigned short* p1) {
    BFrag f;
    f.q[0] = *(const u32x4*)p0;
    f.q[1] = *(const u32x4*)p1;
    return f;
}

__device__ __forceinline__ v8f vzero8() {
    v8f z = {0.f, 0.f, 0.f, 0.f, 0.f, 0.f, 0.f, 0.f};
    return z;
}

// ---------------- conversion kernels ----------------

__global__ __launch_bounds__(256) void convert_f32_bf16(
    const float* __restrict__ src, unsigned short* __restrict__ dst, int n) {
    int i = blockIdx.x * 256 + threadIdx.x;
    if (i < n) dst[i] = f2bf(src[i]);
}

// W[K][C] f32 (row-major)  ->  Wt[C][K] bf16 (row-major)
__global__ __launch_bounds__(256) void transpose_bf16(
    const float* __restrict__ w, unsigned short* __restrict__ wt, int K, int C) {
    int i = blockIdx.x * 256 + threadIdx.x;
    if (i >= K * C) return;
    int n = i / K;            // output row  (column of W)
    int e = i - n * K;        // output col  (row of W)
    wt[i] = f2bf(w[(size_t)e * C + n]);
}

// ---------------- QKV GEMM:  kqv = x @ Wqkv + b  ----------------
// A = xb [BS_, E_] bf16 row-major ; B^T = wt [N3_, E_] bf16 row-major.
// Each wave: 32(M) x 64(N) tile.  Block = 8 waves = 64(M) x 256(N).
__global__ __launch_bounds__(256) void qkv_gemm(
    const unsigned short* __restrict__ xb, const unsigned short* __restrict__ wt,
    const float* __restrict__ bias,
    unsigned short* __restrict__ kh, unsigned short* __restrict__ qh,
    unsigned short* __restrict__ vt) {
    const int lane = threadIdx.x & 31, w = threadIdx.x >> 5;
    const int lm = lane & 15, half = lane >> 4;
    const int M0 = blockIdx.y * 64 + (w & 1) * 32;
    const int N0 = blockIdx.x * 256 + (w >> 1) * 64;
    const int kl = half * 16, kh2 = half * 16 + 8;       // A-frag k offsets

    v8f acc[8];
#pragma unroll
    for (int i = 0; i < 8; ++i) acc[i] = vzero8();

    const unsigned short* a0base = xb + (size_t)(M0 + lm) * E_;
    const unsigned short* a1base = xb + (size_t)(M0 + 16 + lm) * E_;

    for (int k0 = 0; k0 < E_; k0 += 32) {
        BFrag a0 = load_frag2(a0base + k0 + kl, a0base + k0 + kh2);
        BFrag a1 = load_frag2(a1base + k0 + kl, a1base + k0 + kh2);
#pragma unroll
        for (int t = 0; t < 4; ++t) {
            const unsigned short* bp =
                wt + (size_t)(N0 + t * 16 + lm) * E_ + k0 + half * 16;
            BFrag b;
            b.q[0] = *(const u32x4*)bp;
            b.q[1] = *(const u32x4*)(bp + 8);
            acc[t] = __builtin_amdgcn_wmma_f32_16x16x32_bf16(
                false, a0.v, false, b.v, (short)0, acc[t], false, false);
            acc[4 + t] = __builtin_amdgcn_wmma_f32_16x16x32_bf16(
                false, a1.v, false, b.v, (short)0, acc[4 + t], false, false);
        }
    }

#pragma unroll
    for (int t = 0; t < 8; ++t) {
        int mBase = M0 + ((t >= 4) ? 16 : 0);
        int n = N0 + (t & 3) * 16 + lm;
        int sec = n >> 10;             // 0:k 1:q 2:v  (torch split order)
        int nr = n & 1023;
        int h = nr >> 6, d = nr & 63;
        float bv = bias[n];
#pragma unroll
        for (int r = 0; r < 8; ++r) {
            int m = mBase + r + 8 * half;
            int b = m >> 11, s = m & 2047;
            float val = acc[t][r] + bv;
            size_t hb = (size_t)(b * H_ + h);
            if (sec == 0)      kh[(hb * S_ + s) * D_ + d] = f2bf(val);
            else if (sec == 1) qh[(hb * S_ + s) * D_ + d] = f2bf(val * 0.125f);
            else               vt[(hb * D_ + d) * S_ + s] = f2bf(val);
        }
    }
}

// ---------------- FlashAttention (causal) ----------------
// One wave handles 16 query rows of one (b,h).  Block = 4 waves.
__global__ __launch_bounds__(128) void flash_attn(
    const unsigned short* __restrict__ qh, const unsigned short* __restrict__ kh,
    const unsigned short* __restrict__ vt, unsigned short* __restrict__ yb) {
    __shared__ __align__(16) unsigned short pl[4][16 * 32];   // per-wave P tile

    const int lane = threadIdx.x & 31, w = threadIdx.x >> 5;
    const int lm = lane & 15, half = lane >> 4;
    const int task = blockIdx.x * 4 + w;
    const int qt = task & 127;        // query tile (S_/16 = 128)
    const int bh = task >> 7;         // batch*head
    const int qs = qt * 16;
    const int kl = half * 16, kh2 = half * 16 + 8;

    const unsigned short* qp = qh + (size_t)bh * S_ * D_;
    const unsigned short* kp = kh + (size_t)bh * S_ * D_;
    const unsigned short* vp = vt + (size_t)bh * D_ * S_;
    unsigned short* myp = pl[w];

    // Q A-fragments for d=0..31 and d=32..63 (loaded once)
    const unsigned short* qrow = qp + (size_t)(qs + lm) * D_;
    BFrag qa0 = load_frag2(qrow + kl, qrow + kh2);
    BFrag qa1 = load_frag2(qrow + 32 + kl, qrow + 32 + kh2);

    v8f o[4];
    float mrow[8], lrow[8];
#pragma unroll
    for (int t = 0; t < 4; ++t) o[t] = vzero8();
#pragma unroll
    for (int r = 0; r < 8; ++r) { mrow[r] = -1.0e30f; lrow[r] = 0.f; }

    const int nb = qt / 2 + 1;        // 32-key blocks covering keys <= qs+15
    for (int jb = 0; jb < nb; ++jb) {
        const int j0 = jb * 32;
        v8f s0 = vzero8(), s1 = vzero8();

        // scores: S = Q(16x64) @ K^T(64x32keys), two 16-key tiles
        const unsigned short* kr0 = kp + (size_t)(j0 + lm) * D_;
        const unsigned short* kr1 = kp + (size_t)(j0 + 16 + lm) * D_;
        {
            BFrag b;
            b.q[0] = *(const u32x4*)(kr0 + half * 16);
            b.q[1] = *(const u32x4*)(kr0 + half * 16 + 8);
            s0 = __builtin_amdgcn_wmma_f32_16x16x32_bf16(false, qa0.v, false, b.v,
                                                         (short)0, s0, false, false);
            b.q[0] = *(const u32x4*)(kr0 + 32 + half * 16);
            b.q[1] = *(const u32x4*)(kr0 + 32 + half * 16 + 8);
            s0 = __builtin_amdgcn_wmma_f32_16x16x32_bf16(false, qa1.v, false, b.v,
                                                         (short)0, s0, false, false);
            b.q[0] = *(const u32x4*)(kr1 + half * 16);
            b.q[1] = *(const u32x4*)(kr1 + half * 16 + 8);
            s1 = __builtin_amdgcn_wmma_f32_16x16x32_bf16(false, qa0.v, false, b.v,
                                                         (short)0, s1, false, false);
            b.q[0] = *(const u32x4*)(kr1 + 32 + half * 16);
            b.q[1] = *(const u32x4*)(kr1 + 32 + half * 16 + 8);
            s1 = __builtin_amdgcn_wmma_f32_16x16x32_bf16(false, qa1.v, false, b.v,
                                                         (short)0, s1, false, false);
        }

        // causal mask (only needed near/after the diagonal)
        if (j0 + 31 >= qs) {
#pragma unroll
            for (int r = 0; r < 8; ++r) {
                int qrowi = qs + r + 8 * half;
                if (j0 + lm > qrowi)      s0[r] = -1.0e30f;
                if (j0 + 16 + lm > qrowi) s1[r] = -1.0e30f;
            }
        }

        // online softmax: rows live in 16-lane halves -> butterfly reductions
#pragma unroll
        for (int r = 0; r < 8; ++r) {
            float mx = fmaxf(s0[r], s1[r]);
            mx = fmaxf(mx, __shfl_xor(mx, 1));
            mx = fmaxf(mx, __shfl_xor(mx, 2));
            mx = fmaxf(mx, __shfl_xor(mx, 4));
            mx = fmaxf(mx, __shfl_xor(mx, 8));
            float mn = fmaxf(mrow[r], mx);
            float al = __expf(mrow[r] - mn);
            mrow[r] = mn;
            float p0 = __expf(s0[r] - mn);
            float p1 = __expf(s1[r] - mn);
            float rs = p0 + p1;
            rs += __shfl_xor(rs, 1);
            rs += __shfl_xor(rs, 2);
            rs += __shfl_xor(rs, 4);
            rs += __shfl_xor(rs, 8);
            lrow[r] = lrow[r] * al + rs;
            o[0][r] *= al; o[1][r] *= al; o[2][r] *= al; o[3][r] *= al;
            int row = r + 8 * half;
            myp[row * 32 + lm]      = f2bf(p0);   // C-layout -> LDS (row-major 16x32)
            myp[row * 32 + 16 + lm] = f2bf(p1);
        }

        // reload P in A-fragment layout (DS ops are in-order per wave)
        BFrag pf = load_frag2(myp + lm * 32 + kl, myp + lm * 32 + kh2);

        // O += P(16x32) @ V(32x64): 4 B-frags from transposed V rows
#pragma unroll
        for (int t = 0; t < 4; ++t) {
            const unsigned short* vr =
                vp + (size_t)(t * 16 + lm) * S_ + j0 + half * 16;
            BFrag vb;
            vb.q[0] = *(const u32x4*)vr;
            vb.q[1] = *(const u32x4*)(vr + 8);
            o[t] = __builtin_amdgcn_wmma_f32_16x16x32_bf16(false, pf.v, false, vb.v,
                                                           (short)0, o[t], false, false);
        }
    }

    // normalize + write back to [B,S,E] bf16
    const int b = bh >> 4, h = bh & 15;
#pragma unroll
    for (int t = 0; t < 4; ++t) {
#pragma unroll
        for (int r = 0; r < 8; ++r) {
            float val = o[t][r] / lrow[r];
            int row = qs + r + 8 * half;
            yb[(size_t)(b * S_ + row) * E_ + h * 64 + t * 16 + lm] = f2bf(val);
        }
    }
}

// ---------------- projection GEMM:  out = y @ Wproj + b  (fp32 out) --------
__global__ __launch_bounds__(256) void proj_gemm(
    const unsigned short* __restrict__ yb, const unsigned short* __restrict__ wt,
    const float* __restrict__ bias, float* __restrict__ out) {
    const int lane = threadIdx.x & 31, w = threadIdx.x >> 5;
    const int lm = lane & 15, half = lane >> 4;
    const int M0 = blockIdx.y * 64 + (w & 1) * 32;
    const int N0 = blockIdx.x * 256 + (w >> 1) * 64;
    const int kl = half * 16, kh2 = half * 16 + 8;

    v8f acc[8];
#pragma unroll
    for (int i = 0; i < 8; ++i) acc[i] = vzero8();

    const unsigned short* a0base = yb + (size_t)(M0 + lm) * E_;
    const unsigned short* a1base = yb + (size_t)(M0 + 16 + lm) * E_;

    for (int k0 = 0; k0 < E_; k0 += 32) {
        BFrag a0 = load_frag2(a0base + k0 + kl, a0base + k0 + kh2);
        BFrag a1 = load_frag2(a1base + k0 + kl, a1base + k0 + kh2);
#pragma unroll
        for (int t = 0; t < 4; ++t) {
            const unsigned short* bp =
                wt + (size_t)(N0 + t * 16 + lm) * E_ + k0 + half * 16;
            BFrag b;
            b.q[0] = *(const u32x4*)bp;
            b.q[1] = *(const u32x4*)(bp + 8);
            acc[t] = __builtin_amdgcn_wmma_f32_16x16x32_bf16(
                false, a0.v, false, b.v, (short)0, acc[t], false, false);
            acc[4 + t] = __builtin_amdgcn_wmma_f32_16x16x32_bf16(
                false, a1.v, false, b.v, (short)0, acc[4 + t], false, false);
        }
    }

#pragma unroll
    for (int t = 0; t < 8; ++t) {
        int mBase = M0 + ((t >= 4) ? 16 : 0);
        int n = N0 + (t & 3) * 16 + lm;
        float bv = bias[n];
#pragma unroll
        for (int r = 0; r < 8; ++r) {
            int m = mBase + r + 8 * half;
            out[(size_t)m * E_ + n] = acc[t][r] + bv;
        }
    }
}

// ---------------- host launch ----------------

extern "C" void kernel_launch(void* const* d_in, const int* in_sizes, int n_in,
                              void* d_out, int out_size, void* d_ws, size_t ws_size,
                              hipStream_t stream) {
    const float* x      = (const float*)d_in[0];   // [4,2048,1024]
    const float* W_qkv  = (const float*)d_in[1];   // [1024,3072]
    const float* b_qkv  = (const float*)d_in[2];   // [3072]
    const float* W_proj = (const float*)d_in[3];   // [1024,1024]
    const float* b_proj = (const float*)d_in[4];   // [1024]
    float* out = (float*)d_out;

    char* ws = (char*)d_ws;
    size_t off = 0;
    auto alloc = [&](size_t bytes) {
        void* p = ws + off;
        off += (bytes + 255) & ~(size_t)255;
        return p;
    };
    unsigned short* xb     = (unsigned short*)alloc((size_t)BS_ * E_ * 2);
    unsigned short* wqkvT  = (unsigned short*)alloc((size_t)N3_ * E_ * 2);
    unsigned short* wprojT = (unsigned short*)alloc((size_t)E_ * E_ * 2);
    unsigned short* kh     = (unsigned short*)alloc((size_t)BS_ * E_ * 2);
    unsigned short* qh     = (unsigned short*)alloc((size_t)BS_ * E_ * 2);
    unsigned short* vt     = (unsigned short*)alloc((size_t)BS_ * E_ * 2);
    unsigned short* yb     = (unsigned short*)alloc((size_t)BS_ * E_ * 2);
    (void)ws_size; (void)in_sizes; (void)n_in; (void)out_size;

    const int nX = BS_ * E_;
    convert_f32_bf16<<<nX / 256, 256, 0, stream>>>(x, xb, nX);
    transpose_bf16<<<(N3_ * E_) / 256, 256, 0, stream>>>(W_qkv, wqkvT, E_, N3_);
    transpose_bf16<<<(E_ * E_) / 256, 256, 0, stream>>>(W_proj, wprojT, E_, E_);

    qkv_gemm<<<dim3(N3_ / 256, BS_ / 64), 256, 0, stream>>>(
        xb, wqkvT, b_qkv, kh, qh, vt);

    flash_attn<<<(B_ * H_ * (S_ / 16)) / 4, 128, 0, stream>>>(qh, kh, vt, yb);

    proj_gemm<<<dim3(E_ / 256, BS_ / 64), 256, 0, stream>>>(
        yb, wprojT, b_proj, out);
}
